// S6Layer_29815662969397
// MI455X (gfx1250) — matrix operand order repaired and checked by reading, not compiled, not run on hardware
//
#include <hip/hip_runtime.h>
#include <math.h>

// ---------------------------------------------------------------------------
// S6 (selective SSM) layer for MI455X / gfx1250, wave32.
//   B=1, L=4096, D=1024, N=16 (hardcoded to match the reference harness).
// Pipeline:
//   k0a: swizzle W_delta/W_B/W_C (f32) into bf16 WMMA B-fragment order
//   k1 : RMSNorm -> xn (f32) + xn (bf16)
//   k2 : delta = soft_clamp(softplus(xn @ W_delta + b)) via v_wmma bf16.
//        A-tile (16x1024 bf16 = 32KB) staged into LDS via TDM
//        (tensor_load_to_lds + s_wait_tensorcnt), B streamed from global.
//   k3 : Bp = xn @ W_B, C = xn @ W_C via v_wmma bf16
//   k4 : chunked scan pass A  (per-chunk prod(A_bar), h-from-zero)
//   k5 : carry combine across chunks (sequential, tiny)
//   k6 : chunked scan pass B  (real states + y with shfl reduction over N)
// ---------------------------------------------------------------------------

typedef __bf16 bf16_t;
typedef bf16_t v16bf __attribute__((ext_vector_type(16)));
typedef float  v8f   __attribute__((ext_vector_type(8)));
typedef unsigned int u32x4 __attribute__((ext_vector_type(4)));
typedef unsigned int u32x8 __attribute__((ext_vector_type(8)));

#define L_SEQ   4096
#define D_DIM   1024
#define N_ST    16
#define KT_D    (D_DIM / 32)         // 32 K-tiles of 32
#define NCHUNK  32
#define TCHUNK  (L_SEQ / NCHUNK)     // 128
#define DN      (D_DIM * N_ST)       // 16384

__device__ __forceinline__ float softplus_f(float v) {
    return (v > 20.0f) ? v : log1pf(expf(v));
}
__device__ __forceinline__ float soft_clamp_f(float v) {
    const float c = 0.5f * (10.0f + 1e-4f);
    const float h = 0.5f * (10.0f - 1e-4f);
    return c + h * tanhf((v - c) / h);
}

// ---------------------------------------------------------------------------
// WMMA bf16 fragments (CDNA5 ISA 7.12.2 layouts, wave32).
// A: 16x32 (MxK) row-major.  lane<16: row=lane,    K in {0..7}u{16..23}
//                            lane>=16: row=lane-16, K in {8..15}u{24..31}
// B: 32x16 (KxN).            lane<16: col=lane,    K in {0..7}u{16..23}
//                            lane>=16: col=lane-16, K in {8..15}u{24..31}
// B fragments are pre-swizzled so each lane's 16 elements are contiguous.
// ---------------------------------------------------------------------------
union FragB16 { v16bf v; unsigned int u[8]; unsigned short h[16]; };

__device__ __forceinline__ v16bf load_a_frag_g(const bf16_t* __restrict__ A,
                                               int lda, int kc, int lane) {
    FragB16 f;
    const int row = lane & 15;
    const int b0  = (lane < 16) ? 0 : 4;   // (kbase0/2) in u32 units
    const unsigned int* p =
        (const unsigned int*)(A + (size_t)row * lda + kc);
#pragma unroll
    for (int i = 0; i < 4; ++i) f.u[i]     = p[b0 + i];       // K = kbase0..+7
#pragma unroll
    for (int i = 0; i < 4; ++i) f.u[4 + i] = p[b0 + 8 + i];   // K = kbase0+16..
    return f.v;
}

// Same fragment gather, but from the LDS-resident A tile (-> ds_load_b128).
__device__ __forceinline__ v16bf load_a_frag_lds(const bf16_t* A,
                                                 int lda, int kc, int lane) {
    FragB16 f;
    const int row = lane & 15;
    const int b0  = (lane < 16) ? 0 : 4;
    const uint4* p = (const uint4*)(A + (size_t)row * lda + kc);
    uint4 lo = p[b0 >> 2];        // 16B at kbase0
    uint4 hi = p[(b0 >> 2) + 2];  // 16B at kbase0+16
    f.u[0] = lo.x; f.u[1] = lo.y; f.u[2] = lo.z; f.u[3] = lo.w;
    f.u[4] = hi.x; f.u[5] = hi.y; f.u[6] = hi.z; f.u[7] = hi.w;
    return f.v;
}

// Pre-swizzled B fragment: 16 contiguous bf16 (32 bytes) per lane.
__device__ __forceinline__ v16bf load_b_frag_swz(const bf16_t* __restrict__ p) {
    FragB16 f;
    const uint4* q = (const uint4*)p;
    uint4 lo = q[0];
    uint4 hi = q[1];
    f.u[0] = lo.x; f.u[1] = lo.y; f.u[2] = lo.z; f.u[3] = lo.w;
    f.u[4] = hi.x; f.u[5] = hi.y; f.u[6] = hi.z; f.u[7] = hi.w;
    return f.v;
}

// ---------------------------------------------------------------------------
// TDM: async-DMA a [rows x D_DIM] bf16 tile from global into LDS.
// Builds a 2-D Tensor DMA Descriptor (D# groups 0+1, ISA 08_async_tensor §8)
// and issues the 2-operand tensor_load_to_lds (VADDR2/3 = NULL => up-to-2D).
// Call from a single wave; completion via s_wait_tensorcnt.
// ---------------------------------------------------------------------------
__device__ __forceinline__ void tdm_load_tile_to_lds(const bf16_t* gsrc,
                                                     void* lds_dst,
                                                     unsigned rows) {
    const unsigned long long ga = (unsigned long long)gsrc;
    const unsigned lds_addr = (unsigned)(unsigned long long)lds_dst;

    u32x4 g0;
    g0[0] = 1u;                                   // count=1, user desc
    g0[1] = lds_addr;                             // lds_addr [63:32]
    g0[2] = (unsigned)ga;                         // global_addr lo
    g0[3] = ((unsigned)(ga >> 32) & 0x01FFFFFFu)  // global_addr [56:32]
            | 0x80000000u;                        // type=2 ("image") [127:126]

    u32x8 g1;
    g1[0] = 1u << 16;                 // workgroup_mask=0, data_size=1 (2B)
    g1[1] = (D_DIM & 0xFFFFu) << 16;  // tensor_dim0[15:0] at [63:48]
    g1[2] = ((unsigned)L_SEQ & 0xFFFFu) << 16;  // dim0 hi=0 | tensor_dim1 lo
    g1[3] = (D_DIM & 0xFFFFu) << 16;  // tensor_dim1 hi=0 | tile_dim0 [127:112]
    g1[4] = rows & 0xFFFFu;           // tile_dim1 [143:128], tile_dim2=0
    g1[5] = D_DIM;                    // tensor_dim0_stride lo [191:160]
    g1[6] = 0u;                       // stride0 hi | tensor_dim1_stride lo
    g1[7] = 0u;                       // tensor_dim1_stride hi (unused, 2D)

    asm volatile("tensor_load_to_lds %0, %1" :: "s"(g0), "s"(g1) : "memory");
}

__device__ __forceinline__ void tdm_wait_all() {
#if __has_builtin(__builtin_amdgcn_s_wait_tensorcnt)
    __builtin_amdgcn_s_wait_tensorcnt(0);
#else
    asm volatile("s_wait_tensorcnt 0x0" ::: "memory");
#endif
}

// ---------------------------------------------------------------------------
// k0a: swizzle W (kdim x ncols, row-major f32) into bf16 fragment order:
//   Wsw[((ct*KT + kt)*32 + lane)*16 + i] = W[(kt*32 + ko + i + 8*(i>=8))*ncols + ct*16 + (lane&15)]
// ---------------------------------------------------------------------------
__global__ void k0_swizzle_b(const float* __restrict__ W,
                             bf16_t* __restrict__ Wsw,
                             int ncols, int kdim) {
    const int gid  = blockIdx.x * blockDim.x + threadIdx.x;
    const int lane = gid & 31;
    const int kt   = (gid >> 5) % (kdim / 32);
    const int ct   = (gid >> 5) / (kdim / 32);
    if (ct >= ncols / 16) return;
    const int col = ct * 16 + (lane & 15);
    const int ko  = (lane < 16) ? 0 : 8;
    bf16_t* dst = Wsw + ((size_t)(ct * (kdim / 32) + kt) * 32 + lane) * 16;
#pragma unroll
    for (int i = 0; i < 16; ++i)
        dst[i] = (bf16_t)W[(size_t)(kt * 32 + ko + i + ((i >= 8) ? 8 : 0)) * ncols + col];
}

// ---------------------------------------------------------------------------
// k1: RMSNorm. One block (256 thr) per row; 4 elems/thread of D=1024.
// ---------------------------------------------------------------------------
__global__ void k1_rmsnorm(const float* __restrict__ x,
                           const float* __restrict__ w,
                           float* __restrict__ xn,
                           bf16_t* __restrict__ xnb) {
    __shared__ float red[256];
    const int row  = blockIdx.x;
    const int base = threadIdx.x * 4;
    const float* xr = x + (size_t)row * D_DIM;

    float4 xv = *(const float4*)(xr + base);
    red[threadIdx.x] = xv.x * xv.x + xv.y * xv.y + xv.z * xv.z + xv.w * xv.w;
    __syncthreads();
#pragma unroll
    for (int off = 128; off > 0; off >>= 1) {
        if (threadIdx.x < off) red[threadIdx.x] += red[threadIdx.x + off];
        __syncthreads();
    }
    const float rinv = rsqrtf(red[0] * (1.0f / (float)D_DIM) + 1e-6f);

    float4 wv = *(const float4*)(w + base);
    float4 o;
    o.x = wv.x * xv.x * rinv;
    o.y = wv.y * xv.y * rinv;
    o.z = wv.z * xv.z * rinv;
    o.w = wv.w * xv.w * rinv;
    *(float4*)(xn + (size_t)row * D_DIM + base) = o;

    bf16_t* ob = xnb + (size_t)row * D_DIM + base;
    ob[0] = (bf16_t)o.x; ob[1] = (bf16_t)o.y;
    ob[2] = (bf16_t)o.z; ob[3] = (bf16_t)o.w;
}

// ---------------------------------------------------------------------------
// k2: delta GEMM.  grid=(D/512, L/16), block=256 (8 waves).
// A tile (16 x 1024 bf16, 32KB) DMA'd to LDS once via TDM, shared by all
// 8 waves.  Each wave: 16x64 output strip (4 col-tiles), A reused 4x.
// Fused bias + softplus + soft_clamp epilogue.
// ---------------------------------------------------------------------------
__global__ void k2_gemm_delta(const bf16_t* __restrict__ xnb,
                              const bf16_t* __restrict__ Wsw,
                              const float* __restrict__ bd,
                              float* __restrict__ delta) {
    __shared__ bf16_t Atile[16 * D_DIM];           // 32 KB of the 320KB WGP LDS

    const int lane = threadIdx.x & 31;
    const int wave = threadIdx.x >> 5;             // 0..7
    const int row0 = blockIdx.y * 16;
    const int ct0  = blockIdx.x * 32 + wave * 4;   // first 16-col tile

    // Wave 0: one TDM descriptor moves the whole A tile; others park at the
    // barrier. TENSORcnt is per-wave, so the issuing wave waits, then all sync.
    if (threadIdx.x < 32) {
        tdm_load_tile_to_lds(xnb + (size_t)row0 * D_DIM, Atile, 16);
        tdm_wait_all();
    }
    __syncthreads();

    const size_t ct_stride = (size_t)KT_D * 32 * 16;   // elems per col-tile

    v8f acc0 = {}, acc1 = {}, acc2 = {}, acc3 = {};
    for (int kt = 0; kt < KT_D; ++kt) {
        v16bf a = load_a_frag_lds(Atile, D_DIM, kt * 32, lane);
        const bf16_t* bb =
            Wsw + ((size_t)(ct0 * KT_D + kt) * 32 + lane) * 16;
        v16bf b0 = load_b_frag_swz(bb);
        v16bf b1 = load_b_frag_swz(bb + ct_stride);
        v16bf b2 = load_b_frag_swz(bb + 2 * ct_stride);
        v16bf b3 = load_b_frag_swz(bb + 3 * ct_stride);
        acc0 = __builtin_amdgcn_wmma_f32_16x16x32_bf16(false, a, false, b0, (short)0, acc0, false, false);
        acc1 = __builtin_amdgcn_wmma_f32_16x16x32_bf16(false, a, false, b1, (short)0, acc1, false, false);
        acc2 = __builtin_amdgcn_wmma_f32_16x16x32_bf16(false, a, false, b2, (short)0, acc2, false, false);
        acc3 = __builtin_amdgcn_wmma_f32_16x16x32_bf16(false, a, false, b3, (short)0, acc3, false, false);
    }

    const int rowb = row0 + ((lane < 16) ? 0 : 8);
    v8f accs[4] = {acc0, acc1, acc2, acc3};
#pragma unroll
    for (int t = 0; t < 4; ++t) {
        const int col  = (ct0 + t) * 16 + (lane & 15);
        const float bias = bd[col];
#pragma unroll
        for (int i = 0; i < 8; ++i) {
            float v = accs[t][i] + bias;
            delta[(size_t)(rowb + i) * D_DIM + col] = soft_clamp_f(softplus_f(v));
        }
    }
}

// ---------------------------------------------------------------------------
// k3: B/C projections. grid=L/16, block=64 (wave0 -> Bp, wave1 -> C).
// ---------------------------------------------------------------------------
__global__ void k3_gemm_bc(const bf16_t* __restrict__ xnb,
                           const bf16_t* __restrict__ Wbsw,
                           const bf16_t* __restrict__ Wcsw,
                           float* __restrict__ Bp,
                           float* __restrict__ Cc) {
    const int lane = threadIdx.x & 31;
    const int wave = threadIdx.x >> 5;                 // 0 or 1
    const int row0 = blockIdx.x * 16;

    const bf16_t* Arow = xnb + (size_t)row0 * D_DIM;
    const bf16_t* Bm   = wave ? Wcsw : Wbsw;
    float* outp        = wave ? Cc : Bp;

    v8f acc = {};
#pragma unroll 4
    for (int kt = 0; kt < KT_D; ++kt) {
        v16bf a = load_a_frag_g(Arow, D_DIM, kt * 32, lane);
        v16bf b = load_b_frag_swz(Bm + ((size_t)kt * 32 + lane) * 16);
        acc = __builtin_amdgcn_wmma_f32_16x16x32_bf16(
            false, a, false, b, (short)0, acc, false, false);
    }

    const int col  = lane & 15;
    const int rowb = row0 + ((lane < 16) ? 0 : 8);
#pragma unroll
    for (int i = 0; i < 8; ++i)
        outp[(size_t)(rowb + i) * N_ST + col] = acc[i];
}

// ---------------------------------------------------------------------------
// Discretization helper (guarded Tustin), recomputed in-register in the scan.
// ---------------------------------------------------------------------------
__device__ __forceinline__ void tustin_step(float dt, float a, float xv,
                                            float bp, float& Ab, float& Bx) {
    float half  = 0.5f * dt * a;
    float dinv  = 1.0f / fmaxf(1.0f - half, 1e-8f);
    Ab = (1.0f + half) * dinv;
    Bx = sqrtf(dt) * bp * dinv * xv;
}

// ---------------------------------------------------------------------------
// k4: scan pass A: per-chunk cumulative A product + h from zero state.
// grid=(D/16, NCHUNK), block=256 (thread = (dloc, n)).
// ---------------------------------------------------------------------------
__global__ void k4_scan_passA(const float* __restrict__ delta,
                              const float* __restrict__ xn,
                              const float* __restrict__ Bp,
                              const float* __restrict__ A_log,
                              float* __restrict__ Pchunk,
                              float* __restrict__ Hchunk) {
    const int n    = threadIdx.x & 15;
    const int dloc = threadIdx.x >> 4;
    const int d    = blockIdx.x * 16 + dloc;
    const int chunk = blockIdx.y;
    const float a  = -expf(A_log[d * N_ST + n]);

    float h = 0.0f, P = 1.0f;
    const int l0 = chunk * TCHUNK;
    for (int t = 0; t < TCHUNK; ++t) {
        const int l = l0 + t;
        __builtin_prefetch(delta + (size_t)(l + 8) * D_DIM + d, 0, 0);
        const float dt = delta[(size_t)l * D_DIM + d];
        const float xv = xn[(size_t)l * D_DIM + d];
        const float bp = Bp[l * N_ST + n];
        float Ab, Bx;
        tustin_step(dt, a, xv, bp, Ab, Bx);
        h = Ab * h + Bx;
        P *= Ab;
    }
    const int dn = d * N_ST + n;
    Pchunk[chunk * DN + dn] = P;
    Hchunk[chunk * DN + dn] = h;
}

// ---------------------------------------------------------------------------
// k5: carry combine across chunks (sequential over NCHUNK per state).
// ---------------------------------------------------------------------------
__global__ void k5_scan_carry(const float* __restrict__ Pchunk,
                              const float* __restrict__ Hchunk,
                              float* __restrict__ Hin) {
    const int dn = blockIdx.x * blockDim.x + threadIdx.x;
    if (dn >= DN) return;
    float h = 0.0f;
#pragma unroll 4
    for (int k = 0; k < NCHUNK; ++k) {
        Hin[k * DN + dn] = h;
        h = Pchunk[k * DN + dn] * h + Hchunk[k * DN + dn];
    }
}

// ---------------------------------------------------------------------------
// k6: scan pass B: real states + output. y_t[d] = sum_n C_t[n]*h_t[d,n]
// reduced with shfl_xor over the 16-lane N group (wave32-safe).
// ---------------------------------------------------------------------------
__global__ void k6_scan_passB(const float* __restrict__ delta,
                              const float* __restrict__ xn,
                              const float* __restrict__ Bp,
                              const float* __restrict__ Cc,
                              const float* __restrict__ A_log,
                              const float* __restrict__ Dskip,
                              const float* __restrict__ Hin,
                              float* __restrict__ y) {
    const int n    = threadIdx.x & 15;
    const int dloc = threadIdx.x >> 4;
    const int d    = blockIdx.x * 16 + dloc;
    const int chunk = blockIdx.y;
    const float a   = -expf(A_log[d * N_ST + n]);
    const float dsk = Dskip[d];

    float h = Hin[chunk * DN + d * N_ST + n];
    const int l0 = chunk * TCHUNK;
    for (int t = 0; t < TCHUNK; ++t) {
        const int l = l0 + t;
        __builtin_prefetch(delta + (size_t)(l + 8) * D_DIM + d, 0, 0);
        __builtin_prefetch(xn + (size_t)(l + 8) * D_DIM + d, 0, 0);
        const float dt = delta[(size_t)l * D_DIM + d];
        const float xv = xn[(size_t)l * D_DIM + d];
        const float bp = Bp[l * N_ST + n];
        const float cv = Cc[l * N_ST + n];
        float Ab, Bx;
        tustin_step(dt, a, xv, bp, Ab, Bx);
        h = Ab * h + Bx;
        float cy = cv * h;
        cy += __shfl_xor(cy, 1, 16);
        cy += __shfl_xor(cy, 2, 16);
        cy += __shfl_xor(cy, 4, 16);
        cy += __shfl_xor(cy, 8, 16);
        if (n == 0)
            y[(size_t)l * D_DIM + d] = cy + dsk * xv;
    }
}

// ---------------------------------------------------------------------------
// Host-side launcher.
// ---------------------------------------------------------------------------
extern "C" void kernel_launch(void* const* d_in, const int* in_sizes, int n_in,
                              void* d_out, int out_size, void* d_ws, size_t ws_size,
                              hipStream_t stream) {
    (void)in_sizes; (void)n_in; (void)out_size; (void)ws_size;

    const float* x      = (const float*)d_in[0];
    const float* A_log  = (const float*)d_in[1];
    const float* W_delta= (const float*)d_in[2];
    const float* b_delta= (const float*)d_in[3];
    const float* W_B    = (const float*)d_in[4];
    const float* W_C    = (const float*)d_in[5];
    const float* D_skip = (const float*)d_in[6];
    const float* norm_w = (const float*)d_in[7];
    float* y = (float*)d_out;

    // Workspace carve-up (all sizes multiples of 256 bytes).
    char* ws = (char*)d_ws;
    size_t off = 0;
    float*  xn    = (float*)(ws + off);  off += (size_t)L_SEQ * D_DIM * 4;   // 16 MB
    bf16_t* xnb   = (bf16_t*)(ws + off); off += (size_t)L_SEQ * D_DIM * 2;   //  8 MB
    bf16_t* Wd_sw = (bf16_t*)(ws + off); off += (size_t)D_DIM * D_DIM * 2;   //  2 MB
    bf16_t* Wb_sw = (bf16_t*)(ws + off); off += (size_t)D_DIM * N_ST * 2;
    bf16_t* Wc_sw = (bf16_t*)(ws + off); off += (size_t)D_DIM * N_ST * 2;
    float*  delta = (float*)(ws + off);  off += (size_t)L_SEQ * D_DIM * 4;   // 16 MB
    float*  Bp    = (float*)(ws + off);  off += (size_t)L_SEQ * N_ST * 4;
    float*  Cc    = (float*)(ws + off);  off += (size_t)L_SEQ * N_ST * 4;
    float*  Pchk  = (float*)(ws + off);  off += (size_t)NCHUNK * DN * 4;     //  2 MB
    float*  Hchk  = (float*)(ws + off);  off += (size_t)NCHUNK * DN * 4;
    float*  Hin   = (float*)(ws + off);  off += (size_t)NCHUNK * DN * 4;

    // k0a: swizzle weights into bf16 WMMA fragment order
    {
        int thr_wd = (D_DIM / 16) * KT_D * 32;   // 65536
        k0_swizzle_b<<<thr_wd / 256, 256, 0, stream>>>(W_delta, Wd_sw, D_DIM, D_DIM);
        int thr_wn = (N_ST / 16) * KT_D * 32;    // 1024
        k0_swizzle_b<<<thr_wn / 256, 256, 0, stream>>>(W_B, Wb_sw, N_ST, D_DIM);
        k0_swizzle_b<<<thr_wn / 256, 256, 0, stream>>>(W_C, Wc_sw, N_ST, D_DIM);
    }

    // k1: RMSNorm
    k1_rmsnorm<<<L_SEQ, 256, 0, stream>>>(x, norm_w, xn, xnb);

    // k2: delta GEMM (WMMA bf16, TDM-staged A tile) + fused activation
    k2_gemm_delta<<<dim3(D_DIM / 512, L_SEQ / 16), 256, 0, stream>>>(xnb, Wd_sw, b_delta, delta);

    // k3: B / C projections (WMMA bf16)
    k3_gemm_bc<<<L_SEQ / 16, 64, 0, stream>>>(xnb, Wb_sw, Wc_sw, Bp, Cc);

    // k4-k6: chunked scan
    k4_scan_passA<<<dim3(D_DIM / 16, NCHUNK), 256, 0, stream>>>(delta, xn, Bp, A_log, Pchk, Hchk);
    k5_scan_carry<<<DN / 256, 256, 0, stream>>>(Pchk, Hchk, Hin);
    k6_scan_passB<<<dim3(D_DIM / 16, NCHUNK), 256, 0, stream>>>(delta, xn, Bp, Cc, A_log, D_skip, Hin, y);
}
